// MoELayer_37580963840355
// MI455X (gfx1250) — compile-verified, matrix operand order checked by reading, not run on hardware
//
#include <hip/hip_runtime.h>
#include <math.h>

// ---------------------------------------------------------------------------
// MoE: out[n,r] = sum_e softmax(x@gate_w)[n,e] * (x @ W_e + b_e)[n,r]
// N = D_IN = D_OUT = 4096, E = 8.  ~1.1 TFLOP vs ~640 MB -> compute bound
// (~1700 FLOP/byte), so run the expert GEMMs on v_wmma_f32_16x16x32_bf16 with
// on-the-fly f32->bf16 conversion. Register-staged LDS double buffering
// overlaps global-load latency with the WMMAs; the gate combine is fused into
// the expert epilogue so no [E,N,D_OUT] intermediate exists.
// ---------------------------------------------------------------------------

typedef __attribute__((ext_vector_type(16))) __bf16 v16bf;
typedef __attribute__((ext_vector_type(8)))  __bf16 v8bf;
typedef __attribute__((ext_vector_type(8)))  float  v8f;
typedef __attribute__((ext_vector_type(2)))  float  f32x2;

#define DIMN   4096
#define DIMK   4096
#define DIMR   4096
#define NEXP   8

#define BM     256   // token rows per block
#define BN     128   // output cols per block
#define BK     32    // K chunk (one WMMA K step)
#define THREADS 512  // 16 wave32s
#define BSTRIDE (BK + 8)   // 40 halves = 80 B rows: 16B-aligned, de-conflicted

#define A_PAIRS ((BM * BK) / (2 * THREADS))   // 8 f32x2 per thread
#define B_PAIRS ((BN * BK) / (2 * THREADS))   // 4 pairs per thread

// Two f32 -> packed bf16x2 (round-to-nearest, ties away) in 3 VALU ops:
// two v_add + one v_perm_b32 selecting the high halves.
__device__ __forceinline__ unsigned pack_bf16x2(float a, float b) {
    union { float f; unsigned u; } x, y;
    x.f = a; y.f = b;
    const unsigned ua = x.u + 0x8000u;
    const unsigned ub = y.u + 0x8000u;
    // result bytes: [0]=ua[2] [1]=ua[3] [2]=ub[2] [3]=ub[3]
    return __builtin_amdgcn_perm(ub, ua, 0x07060302u);
}

// Per-lane 16-bit A/B fragment: two contiguous 16-byte LDS loads.
// Element j of the v16bf holds K = (j%8) + 8*(lane/16) + 16*(j/8).
__device__ __forceinline__ v16bf load_frag(const unsigned short* row, int hi) {
    v8bf lo = *(const v8bf*)(const void*)(row + 8 * hi);
    v8bf hh = *(const v8bf*)(const void*)(row + 16 + 8 * hi);
    return __builtin_shufflevector(lo, hh,
        0, 1, 2, 3, 4, 5, 6, 7, 8, 9, 10, 11, 12, 13, 14, 15);
}

// ---------------------------------------------------------------------------
// Kernel 1: gate[n, 0..7] = softmax(x[n,:] @ gate_w + gate_b)
// ---------------------------------------------------------------------------
__global__ __launch_bounds__(128)
void moe_gate_kernel(const float* __restrict__ x,
                     const float* __restrict__ gate_w,
                     const float* __restrict__ gate_b,
                     float* __restrict__ gate) {
    const int n   = blockIdx.x;
    const int tid = threadIdx.x;

    float p[NEXP];
#pragma unroll
    for (int e = 0; e < NEXP; ++e) p[e] = 0.0f;

    const float* xr = x + (size_t)n * DIMK;
    for (int d = tid; d < DIMK; d += 128) {
        const float xv = xr[d];
        const float* gw = gate_w + (size_t)d * NEXP;
#pragma unroll
        for (int e = 0; e < NEXP; ++e) p[e] += xv * gw[e];
    }

    __shared__ float red[128][NEXP];
#pragma unroll
    for (int e = 0; e < NEXP; ++e) red[tid][e] = p[e];
    __syncthreads();
    for (int s = 64; s > 0; s >>= 1) {
        if (tid < s) {
#pragma unroll
            for (int e = 0; e < NEXP; ++e) red[tid][e] += red[tid + s][e];
        }
        __syncthreads();
    }
    if (tid == 0) {
        float z[NEXP], mx = -3.4e38f;
#pragma unroll
        for (int e = 0; e < NEXP; ++e) {
            z[e] = red[0][e] + gate_b[e];
            mx = fmaxf(mx, z[e]);
        }
        float sum = 0.0f;
#pragma unroll
        for (int e = 0; e < NEXP; ++e) { z[e] = __expf(z[e] - mx); sum += z[e]; }
        const float inv = 1.0f / sum;
#pragma unroll
        for (int e = 0; e < NEXP; ++e) gate[(size_t)n * NEXP + e] = z[e] * inv;
    }
}

// ---------------------------------------------------------------------------
// Kernel 2: fused expert GEMMs + gated combine.
// Block: 256x128 out tile, 16 wave32s; wave w: rows 32*(w%8), cols 64*(w/8);
// each wave owns a 2x4 grid of 16x16 WMMA C tiles.
// Pipeline per K-step: global loads (next) -> WMMAs (cur) -> cvt+ds_store
// (next buffer) -> barrier.
// ---------------------------------------------------------------------------
__global__ __launch_bounds__(THREADS)
void moe_expert_kernel(const float* __restrict__ x,
                       const float* __restrict__ expert_w,
                       const float* __restrict__ expert_b,
                       const float* __restrict__ gate,
                       float* __restrict__ out) {
    __shared__ alignas(16) unsigned short Alds[2][BM][BK];       // 2 x 16 KB
    __shared__ alignas(16) unsigned short Blds[2][BN][BSTRIDE];  // 2 x 10 KB

    const int tid     = threadIdx.x;
    const int lane    = tid & 31;
    const int wid     = tid >> 5;        // 0..15
    const int wm      = wid & 7;         // row group: 8 x 32 rows
    const int wn      = wid >> 3;        // col group: 2 x 64 cols
    const int ln      = lane & 15;
    const int hi      = lane >> 4;       // K-half select per ISA layout
    const int rowBase = blockIdx.x * BM;
    const int colBase = blockIdx.y * BN;

    // Loop-invariant staging coordinates (i-strides are compile-time consts):
    const int aM = tid >> 4;             // row 0..31, +32 per i
    const int aK = (tid & 15) << 1;      // even k column
    const int bN = tid & (BN - 1);       // col within tile
    const int bK = (tid >> 7) << 1;      // even k row, +8 per i

    const float* xA = x + (size_t)(rowBase + aM) * DIMK + aK;

    f32x2 aReg[A_PAIRS];
    float bReg[2 * B_PAIRS];

    v8f oacc[2][4];
    const v8f vzero = {};
#pragma unroll
    for (int tm = 0; tm < 2; ++tm)
#pragma unroll
        for (int tn = 0; tn < 4; ++tn) oacc[tm][tn] = vzero;

    for (int e = 0; e < NEXP; ++e) {
        const float* We = expert_w + (size_t)e * DIMK * DIMR;
        const float* wB = We + (size_t)bK * DIMR + colBase + bN;

        // --- pipeline helpers ---
        auto load_stage = [&](int ks) {
#pragma unroll
            for (int i = 0; i < A_PAIRS; ++i)
                aReg[i] = *(const f32x2*)(xA + (size_t)(32 * i) * DIMK + ks);
#pragma unroll
            for (int i = 0; i < B_PAIRS; ++i) {
                const float* p = wB + (size_t)(8 * i + ks) * DIMR;
                bReg[2 * i]     = p[0];
                bReg[2 * i + 1] = p[DIMR];
            }
            // pull the chunk after next toward L2 (global_prefetch_b8)
            __builtin_prefetch(wB + (size_t)(ks + BK) * DIMR, 0, 1);
        };
        auto store_stage = [&](int buf) {
#pragma unroll
            for (int i = 0; i < A_PAIRS; ++i)
                *(unsigned*)&Alds[buf][aM + 32 * i][aK] =
                    pack_bf16x2(aReg[i].x, aReg[i].y);
#pragma unroll
            for (int i = 0; i < B_PAIRS; ++i)
                *(unsigned*)&Blds[buf][bN][bK + 8 * i] =
                    pack_bf16x2(bReg[2 * i], bReg[2 * i + 1]);
        };

        v8f acc[2][4];
#pragma unroll
        for (int tm = 0; tm < 2; ++tm)
#pragma unroll
            for (int tn = 0; tn < 4; ++tn) acc[tm][tn] = vzero;

        // --- prologue: stage first chunk ---
        load_stage(0);
        store_stage(0);
        __syncthreads();

        for (int k0 = 0; k0 < DIMK; k0 += BK) {
            const int cur = (k0 >> 5) & 1;
            const int nxt = cur ^ 1;
            const bool hasNext = (k0 + BK) < DIMK;

            // issue next chunk's global loads first (latency overlapped by
            // the WMMAs below)
            if (hasNext) load_stage(k0 + BK);

            // fragments + 8 WMMAs on current buffer
            v16bf afrag[2];
#pragma unroll
            for (int tm = 0; tm < 2; ++tm)
                afrag[tm] =
                    load_frag(&Alds[cur][wm * 32 + tm * 16 + ln][0], hi);
            v16bf bfrag[4];
#pragma unroll
            for (int tn = 0; tn < 4; ++tn)
                bfrag[tn] =
                    load_frag(&Blds[cur][wn * 64 + tn * 16 + ln][0], hi);

#pragma unroll
            for (int tm = 0; tm < 2; ++tm)
#pragma unroll
                for (int tn = 0; tn < 4; ++tn)
                    acc[tm][tn] = __builtin_amdgcn_wmma_f32_16x16x32_bf16(
                        false, afrag[tm], false, bfrag[tn],
                        (short)0, acc[tm][tn], false, false);

            // convert + park next chunk in the other buffer
            if (hasNext) store_stage(nxt);
            __syncthreads();
        }

        // --- fold expert into output: oacc += g[row] * (acc + b[col]) ---
        float g[2][8];
#pragma unroll
        for (int tm = 0; tm < 2; ++tm) {
            const int r0 = rowBase + wm * 32 + tm * 16 + hi * 8;
#pragma unroll
            for (int r = 0; r < 8; ++r)
                g[tm][r] = gate[(size_t)(r0 + r) * NEXP + e];
        }
        float bcol[4];
#pragma unroll
        for (int tn = 0; tn < 4; ++tn)
            bcol[tn] = expert_b[(size_t)e * DIMR + colBase + wn * 64 +
                                tn * 16 + ln];
#pragma unroll
        for (int tm = 0; tm < 2; ++tm)
#pragma unroll
            for (int tn = 0; tn < 4; ++tn)
#pragma unroll
                for (int r = 0; r < 8; ++r)
                    oacc[tm][tn][r] += g[tm][r] * (acc[tm][tn][r] + bcol[tn]);
    }

    // --- writeback (C layout: VGPR r -> row base + 8*hi + r, col = ln) ---
#pragma unroll
    for (int tm = 0; tm < 2; ++tm)
#pragma unroll
        for (int tn = 0; tn < 4; ++tn) {
            const int row0 = rowBase + wm * 32 + tm * 16 + hi * 8;
            const int col  = colBase + wn * 64 + tn * 16 + ln;
#pragma unroll
            for (int r = 0; r < 8; ++r)
                out[(size_t)(row0 + r) * DIMR + col] = oacc[tm][tn][r];
        }
}

// ---------------------------------------------------------------------------
extern "C" void kernel_launch(void* const* d_in, const int* in_sizes, int n_in,
                              void* d_out, int out_size, void* d_ws,
                              size_t ws_size, hipStream_t stream) {
    const float* x        = (const float*)d_in[0];
    const float* gate_w   = (const float*)d_in[1];
    const float* gate_b   = (const float*)d_in[2];
    const float* expert_w = (const float*)d_in[3];
    const float* expert_b = (const float*)d_in[4];
    float*       out      = (float*)d_out;
    float*       gate     = (float*)d_ws;   // 4096 * 8 f32 = 128 KB scratch

    moe_gate_kernel<<<DIMN, 128, 0, stream>>>(x, gate_w, gate_b, gate);

    dim3 grid(DIMN / BM, DIMR / BN);  // (16, 32): row tiles fastest ->
                                      // consecutive blocks share a weight
                                      // column stripe (L2 reuse)
    moe_expert_kernel<<<grid, THREADS, 0, stream>>>(x, expert_w, expert_b,
                                                    gate, out);
}
